// SpectralRegressor_16879221473308
// MI455X (gfx1250) — matrix operand order repaired and checked by reading, not compiled, not run on hardware
//
#include <hip/hip_runtime.h>
#include <math.h>

// ---------------------------------------------------------------------------
// CDNA5 (gfx1250) FNO SpectralRegressor, round 2.
// - truncated-mode DFTs as bf16 WMMA GEMMs (fp32 accum)
// - ALL intermediates bf16 (halves HBM traffic; workload is bandwidth-bound)
// - contraction-pair-interleaved layouts => fragments load as raw b32 words
// - twiddles / weights pre-expanded per call into fragment-ordered bf16 tables
// ---------------------------------------------------------------------------

typedef __bf16 bf16;
typedef unsigned int u32;
typedef __attribute__((ext_vector_type(16))) __bf16 v16bf;
typedef __attribute__((ext_vector_type(8)))  float  v8f;

#define TWO_PI 6.28318530717958647692f

__device__ __forceinline__ v8f wmma_bf16(v16bf a, v16bf b, v8f c) {
    return __builtin_amdgcn_wmma_f32_16x16x32_bf16(false, a, false, b,
                                                   (short)0, c, false, false);
}

// 16-bit A fragment (16x32): lane m = lane&15, half = lane>>4.
__device__ __forceinline__ int a_k(int i, int half) {
    return 2 * i + 16 * half;
}

// Fragment viewed as 8 raw dwords (each dword = bf16 pair (k, k+1)).
struct Frag { union { v16bf bf; u32 u[8]; }; };

__device__ __forceinline__ u32 pack2(float lo, float hi) {
    unsigned short l = __builtin_bit_cast(unsigned short, (bf16)lo);
    unsigned short h = __builtin_bit_cast(unsigned short, (bf16)hi);
    return ((u32)h << 16) | (u32)l;
}

// ---------------------------------------------------------------------------
// Per-call table prep (deterministic, tiny).  All twiddle tables are stored in
// exact A-fragment order: [set][lane(32)][16 bf16] -> loaded as one v16bf.
// ---------------------------------------------------------------------------
__global__ void k_init_tw(bf16* __restrict__ TZ, bf16* __restrict__ TM,
                          bf16* __restrict__ TI, bf16* __restrict__ TC,
                          bf16* __restrict__ TB1, bf16* __restrict__ TLW,
                          const float* __restrict__ rw1,
                          const float* __restrict__ lw0,
                          const float* __restrict__ lw1) {
    const int tid = threadIdx.x;
    // ---- TZ: rfft-Z A frags [kt2][lane][16]: rows 0..7 cos, 8..15 -sin ----
    for (int idx = tid; idx < 1024; idx += 1024) {
        int el = idx & 15, i = el >> 1, e = el & 1;
        int lane = (idx >> 4) & 31, kt = idx >> 9;
        int m = lane & 15, half = lane >> 4;
        int mode = (m < 8) ? m : (m - 8);
        int z = kt * 32 + a_k(i, half) + e;
        float ang = TWO_PI * (float)(mode * z) * (1.0f / 64.0f);
        TZ[idx] = (bf16)((m >= 8) ? -sinf(ang) : cosf(ang));
    }
    // ---- TM: fwd DFT A frags [var3][kt2][lane][16]: cos, sin, -sin --------
    for (int idx = tid; idx < 3072; idx += 1024) {
        int el = idx & 15, i = el >> 1, e = el & 1;
        int lane = (idx >> 4) & 31, kt = (idx >> 9) & 1, var = idx >> 10;
        int m = lane & 15, half = lane >> 4;
        int f = (m < 8) ? m : (m - 16);
        int y = kt * 32 + a_k(i, half) + e;
        float ang = -TWO_PI * (float)(f * y) * (1.0f / 64.0f);
        float v = (var == 0) ? cosf(ang) : ((var == 1) ? sinf(ang) : -sinf(ang));
        TM[idx] = (bf16)v;
    }
    // ---- TI: inverse DFT A frags [mt4][var2][lane][16] --------------------
    for (int idx = tid; idx < 4096; idx += 1024) {
        int el = idx & 15, i = el >> 1, e = el & 1;
        int lane = (idx >> 4) & 31, var = (idx >> 9) & 1, mt = idx >> 10;
        int m = lane & 15, half = lane >> 4;
        int x = mt * 16 + m;
        int j = a_k(i, half) + e;
        int k = j >> 1, p = j & 1;
        int f = (k < 8) ? k : (k - 16);
        float ang = TWO_PI * (float)(f * x) * (1.0f / 64.0f);
        float v = (var == 0) ? (p ? -sinf(ang) : cosf(ang))
                             : (p ?  cosf(ang) : sinf(ang));
        TI[idx] = (bf16)v;
    }
    // ---- TC: hermitian irfft-Z A frags [mt4][lane][16] --------------------
    for (int idx = tid; idx < 2048; idx += 1024) {
        int el = idx & 15, i = el >> 1, e = el & 1;
        int lane = (idx >> 4) & 31, mt = idx >> 9;
        int m = lane & 15, half = lane >> 4;
        int z = mt * 16 + m;
        int j = a_k(i, half) + e;
        int k = j >> 1, p = j & 1;
        float ang = TWO_PI * (float)(k * z) * (1.0f / 64.0f);
        float v = p ? ((k == 0) ? 0.f : -2.f * sinf(ang))
                    : ((k == 0) ? 1.f :  2.f * cosf(ang));
        TC[idx] = (bf16)v;
    }
    // ---- TB1: rw1 pre-swizzled B frags [nt12][pair16][n16][2] -------------
    for (int idx = tid; idx < 6144; idx += 1024) {
        int e = idx & 1, n = (idx >> 1) & 15, pair = (idx >> 5) & 15, nt = idx >> 9;
        TB1[idx] = (bf16)rw1[(pair * 2 + e) * 192 + nt * 16 + n];
    }
    // ---- TLW: lw0/lw1 B frags [layer2][nt2][pair16][n16][2] ---------------
    for (int idx = tid; idx < 2048; idx += 1024) {
        int e = idx & 1, n = (idx >> 1) & 15, pair = (idx >> 5) & 15;
        int nt = (idx >> 9) & 1, l = idx >> 10;
        const float* lw = l ? lw1 : lw0;
        TLW[idx] = (bf16)lw[(pair * 2 + e) * 32 + nt * 16 + n];
    }
}

// Expand spectral weights into the real-embedded bf16 B' table:
// TW[mode][jpair=ci][n=2co+pb][pa] with (pa,pb): (0,0)=wr (0,1)=wi (1,0)=-wi (1,1)=wr
__global__ void k_prep_w(const float* __restrict__ w1, const float* __restrict__ w2,
                         const float* __restrict__ w3, const float* __restrict__ w4,
                         bf16* __restrict__ TW) {
    const int tid = blockIdx.x * 256 + threadIdx.x;   // 2M threads
    const int mode = tid & 2047;
    const int cc = tid >> 11;
    const int ci = cc >> 5, co = cc & 31;
    const int kz = mode & 7, ky = (mode >> 3) & 15, kx = mode >> 7;
    const float* wt = (kx < 8) ? ((ky < 8) ? w1 : w3) : ((ky < 8) ? w2 : w4);
    const long wi0 = (long)ci * 32768 + (long)co * 1024
                   + (kx & 7) * 128 + (ky & 7) * 16 + kz * 2;
    const float wr = wt[wi0], wi = wt[wi0 + 1];
    const long base = (long)mode * 4096 + ci * 128 + co * 4;
    TW[base + 0] = (bf16)wr;     // pa=0 pb=0
    TW[base + 1] = (bf16)(-wi);  // pa=1 pb=0
    TW[base + 2] = (bf16)wi;     // pa=0 pb=1
    TW[base + 3] = (bf16)wr;     // pa=1 pb=1
}

// ---------------------------------------------------------------------------
// rfft over Z (modes 0..7).  One wave per (b,x,y) slab [64 z x 32 c].
// Out G1r/G1i: [bx256][ypair32][kz8][c32][yparity2] (pair-interleaved over Y).
// ---------------------------------------------------------------------------
template<typename T>
__global__ void k_dft_z(const T* __restrict__ in, bf16* __restrict__ G1r,
                        bf16* __restrict__ G1i, const bf16* __restrict__ TT) {
    const int w = threadIdx.x >> 5, lane = threadIdx.x & 31;
    const int slab = blockIdx.x * 4 + w;              // b*4096 + x*64 + y
    const T* src = in + (long)slab * 2048;
    const v16bf* Tt = (const v16bf*)TT;
    const v16bf a0 = Tt[lane], a1 = Tt[32 + lane];
    const int half = lane >> 4, nl = lane & 15;
    const int bx = slab >> 6, y = slab & 63;
    bf16* dst = half ? G1i : G1r;
    const long obase = (long)bx * 16384 + (long)(y >> 1) * 512 + (y & 1);
#pragma unroll
    for (int nt = 0; nt < 2; ++nt) {
        const int n = nt * 16 + nl;
        Frag b0, b1;
#pragma unroll
        for (int i = 0; i < 8; ++i) {
            const int k0 = 2 * i + 16 * half;
#pragma unroll
            for (int e = 0; e < 2; ++e) {
                b0.bf[2 * i + e] = (bf16)src[(k0 + e) * 32 + n];
                b1.bf[2 * i + e] = (bf16)src[(32 + k0 + e) * 32 + n];
            }
        }
        v8f acc = {};
        acc = wmma_bf16(a0, b0.bf, acc);
        acc = wmma_bf16(a1, b1.bf, acc);
#pragma unroll
        for (int r = 0; r < 8; ++r)
            dst[obase + r * 64 + n * 2] = (bf16)acc[r];
    }
}

// ---------------------------------------------------------------------------
// Forward truncated complex DFT over one 64-axis -> 16 modes.
// In: split R/I planes, pair-interleaved along contraction axis (8 b32/frag).
// Out: CPLX=true  -> complex-interleaved single array (packed b32 stores)
//      CPLX=false -> split planes pair-interleaved along this slab's i1 axis.
// ---------------------------------------------------------------------------
template<bool CPLX>
__global__ void k_dft_mode(const bf16* __restrict__ inR, const bf16* __restrict__ inI,
                           bf16* __restrict__ outR, bf16* __restrict__ outI,
                           const bf16* __restrict__ TT,
                           int n1, int n2,
                           long si0, long si1, long si2, long siP,
                           long so0, long so1, long so2, long soM) {
    const int w = threadIdx.x >> 5, lane = threadIdx.x & 31;
    const int slab = blockIdx.x * 4 + w;
    const int i2 = slab % n2;
    const int t  = slab / n2;
    const int i1 = t % n1;
    const int i0 = t / n1;
    const long ibase = i0 * si0 + i1 * si1 + i2 * si2;
    const int half = lane >> 4, nl = lane & 15;
    const v16bf* Tt = (const v16bf*)TT;
    const v16bf ar0 = Tt[lane],       ar1 = Tt[32 + lane];
    const v16bf ai0 = Tt[64 + lane],  ai1 = Tt[96 + lane];
    const v16bf an0 = Tt[128 + lane], an1 = Tt[160 + lane];
    long obase;
    if (CPLX) obase = i0 * so0 + (long)i1 * so1 + i2 * so2;
    else      obase = i0 * so0 + (long)(i1 >> 1) * so1 + i2 * so2 + (i1 & 1);
#pragma unroll
    for (int nt = 0; nt < 2; ++nt) {
        const int n2e = (nt * 16 + nl) * 2;
        Frag br0, br1, bi0, bi1;
#pragma unroll
        for (int i = 0; i < 8; ++i) {
            const long p0 = ibase + (long)(i + 8 * half) * siP + n2e;
            const long p1 = ibase + (long)(16 + i + 8 * half) * siP + n2e;
            br0.u[i] = *(const u32*)(inR + p0);
            br1.u[i] = *(const u32*)(inR + p1);
            bi0.u[i] = *(const u32*)(inI + p0);
            bi1.u[i] = *(const u32*)(inI + p1);
        }
        v8f accR = {}, accI = {};
        accR = wmma_bf16(ar0, br0.bf, accR);   // Wr*Gr
        accR = wmma_bf16(ar1, br1.bf, accR);
        accR = wmma_bf16(an0, bi0.bf, accR);   // - Wi*Gi
        accR = wmma_bf16(an1, bi1.bf, accR);
        accI = wmma_bf16(ar0, bi0.bf, accI);   // Wr*Gi
        accI = wmma_bf16(ar1, bi1.bf, accI);
        accI = wmma_bf16(ai0, br0.bf, accI);   // + Wi*Gr
        accI = wmma_bf16(ai1, br1.bf, accI);
#pragma unroll
        for (int r = 0; r < 8; ++r) {
            const int mode = r + 8 * half;
            const long o = obase + (long)mode * soM + n2e;
            if (CPLX) {
                *(u32*)(outR + o) = pack2(accR[r], accI[r]);
            } else {
                outR[o] = (bf16)accR[r];
                outI[o] = (bf16)accI[r];
            }
        }
    }
}

// ---------------------------------------------------------------------------
// Spectral channel mix: per mode, complex [B=4 x 32] @ [32 x 32] via the 2x2
// real-block embedding (K=64, N=64).  A from complex-interleaved G3 (b32
// loads); B from the pre-expanded TW table (coalesced b32 loads, no LDS).
// ---------------------------------------------------------------------------
__global__ void k_spec(const bf16* __restrict__ G3, const bf16* __restrict__ TW,
                       bf16* __restrict__ H) {
    const int w = threadIdx.x >> 5, lane = threadIdx.x & 31;
    const int slab = blockIdx.x * 4 + w;              // mode id
    const int half = lane >> 4, ml = lane & 15;
    const int kz = slab & 7, ky = (slab >> 3) & 15, kx = slab >> 7;
    const long gbase = (long)kx * 8192 + ky * 512 + kz * 64;
    const long wbase = (long)slab * 4096;
    Frag a0, a1;
#pragma unroll
    for (int i = 0; i < 8; ++i) {
        const int ap = i + 8 * half;   // a_k>>1 = ci
        if (ml < 4) {
            const long r0 = (long)ml * 131072 + gbase;
            a0.u[i] = *(const u32*)(G3 + r0 + (long)ap * 2);
            a1.u[i] = *(const u32*)(G3 + r0 + (long)(16 + ap) * 2);
        } else { a0.u[i] = 0; a1.u[i] = 0; }
    }
#pragma unroll
    for (int nt = 0; nt < 4; ++nt) {
        Frag b0, b1;
#pragma unroll
        for (int i = 0; i < 8; ++i) {
            const int bp = i + 8 * half;
            const long c2 = (nt * 16 + ml) * 2;
            b0.u[i] = *(const u32*)(TW + wbase + (long)bp * 128 + c2);
            b1.u[i] = *(const u32*)(TW + wbase + (long)(16 + bp) * 128 + c2);
        }
        v8f acc = {};
        acc = wmma_bf16(a0.bf, b0.bf, acc);
        acc = wmma_bf16(a1.bf, b1.bf, acc);
        if (half == 0) {                                // rows 0..3 = batch
            const int n = nt * 16 + ml, co = n >> 1, pb = n & 1;
#pragma unroll
            for (int r = 0; r < 4; ++r)
                H[(long)r * 131072 + gbase + co * 2 + pb] = (bf16)acc[r];
        }
    }
}

// ---------------------------------------------------------------------------
// Inverse truncated DFT over one axis: 16 complex modes -> 64 positions.
// Complex-interleaved in and out; K=32 exactly (re/im interleaved contraction).
// ---------------------------------------------------------------------------
__global__ void k_idft(const bf16* __restrict__ in, bf16* __restrict__ out,
                       const bf16* __restrict__ TT,
                       int n1, int n2,
                       long si0, long si1, long si2, long siP,
                       long so0, long so1, long so2, long soM) {
    const int w = threadIdx.x >> 5, lane = threadIdx.x & 31;
    const int slab = blockIdx.x * 4 + w;
    const int i2 = slab % n2;
    const int t  = slab / n2;
    const int i1 = t % n1;
    const int i0 = t / n1;
    const long ibase = i0 * si0 + i1 * si1 + i2 * si2;
    const int half = lane >> 4, nl = lane & 15;
    const v16bf* Tt = (const v16bf*)TT;
    Frag bAll[2];
#pragma unroll
    for (int nt = 0; nt < 2; ++nt)
#pragma unroll
        for (int i = 0; i < 8; ++i)
            bAll[nt].u[i] = *(const u32*)(in + ibase + (long)(i + 8 * half) * siP
                                             + (nt * 16 + nl) * 2);
    const long obase = i0 * so0 + i1 * so1 + i2 * so2;
#pragma unroll
    for (int mt = 0; mt < 4; ++mt) {
        const v16bf aR = Tt[(mt * 2 + 0) * 32 + lane];
        const v16bf aI = Tt[(mt * 2 + 1) * 32 + lane];
#pragma unroll
        for (int nt = 0; nt < 2; ++nt) {
            v8f accR = {}, accI = {};
            accR = wmma_bf16(aR, bAll[nt].bf, accR);
            accI = wmma_bf16(aI, bAll[nt].bf, accI);
            const long c2 = (nt * 16 + nl) * 2;
#pragma unroll
            for (int r = 0; r < 8; ++r) {
                const int xx = mt * 16 + r + 8 * half;
                *(u32*)(out + obase + (long)xx * soM + c2) = pack2(accR[r], accI[r]);
            }
        }
    }
}

// ---------------------------------------------------------------------------
// Fused: hermitian irfft-Z (K=32 GEMM from complex-interleaved I2) + residual
// x@lw+lb GEMM + SiLU.  1/64^3 irfftn scale applied in fp32 post-accum.
// ---------------------------------------------------------------------------
template<typename T>
__global__ void k_combine(const T* actin, const bf16* __restrict__ I2,
                          const bf16* __restrict__ lwtab, const float* __restrict__ lb,
                          bf16* actout, const bf16* __restrict__ TC) {
    const int w = threadIdx.x >> 5, lane = threadIdx.x & 31;
    const int slab = blockIdx.x * 4 + w;              // b*4096 + x*64 + y
    const int half = lane >> 4, nl = lane & 15;
    const long hbase = (long)slab * 512;
    Frag bz[2], blw[2];
#pragma unroll
    for (int nt = 0; nt < 2; ++nt)
#pragma unroll
        for (int i = 0; i < 8; ++i) {
            const int pair = i + 8 * half;
            bz[nt].u[i]  = *(const u32*)(I2 + hbase + (long)pair * 64
                                            + (nt * 16 + nl) * 2);
            blw[nt].u[i] = *(const u32*)(lwtab + ((nt * 16 + pair) * 16 + nl) * 2);
        }
    const v16bf* Tc = (const v16bf*)TC;
    const float scale = 1.0f / 262144.0f;             // 1/64^3
    const long abase = (long)slab * 2048;
#pragma unroll
    for (int mt = 0; mt < 4; ++mt) {
        const v16bf az = Tc[mt * 32 + lane];
        const int z = mt * 16 + nl;
        Frag ar;
#pragma unroll
        for (int i = 0; i < 8; ++i) {
            const int k0 = a_k(i, half);
            ar.bf[2 * i]     = (bf16)actin[abase + (long)z * 32 + k0];
            ar.bf[2 * i + 1] = (bf16)actin[abase + (long)z * 32 + k0 + 1];
        }
#pragma unroll
        for (int nt = 0; nt < 2; ++nt) {
            v8f accS = {}, accR = {};
            accS = wmma_bf16(az, bz[nt].bf, accS);
            accR = wmma_bf16(ar.bf, blw[nt].bf, accR);
            const int c = nt * 16 + nl;
            const float bias = lb[c];
#pragma unroll
            for (int r = 0; r < 8; ++r) {
                const int zz = mt * 16 + r + 8 * half;
                float v = accS[r] * scale + accR[r] + bias;
                actout[abase + (long)zz * 32 + c] = (bf16)(v / (1.f + __expf(-v)));
            }
        }
    }
}

// ---------------------------------------------------------------------------
// Fused MLP: out = silu(x@rw1+rb1)@rw2+rb2.  Hidden [16x192] stays in regs;
// rw1 pre-swizzled (TB1) so B frags are 8 coalesced b32 loads.
// ---------------------------------------------------------------------------
__global__ void k_mlp(const bf16* __restrict__ act, const bf16* __restrict__ TB1,
                      const float* __restrict__ rb1, const float* __restrict__ rw2,
                      const float* __restrict__ rb2, float* __restrict__ out) {
    const int w = threadIdx.x >> 5, lane = threadIdx.x & 31;
    const long row0 = ((long)blockIdx.x * 4 + w) * 16;
    const int half = lane >> 4, nl = lane & 15;
    Frag a;
#pragma unroll
    for (int i = 0; i < 8; ++i)
        a.u[i] = *(const u32*)(act + (row0 + nl) * 32 + a_k(i, half));
    float partial[8] = {0.f, 0.f, 0.f, 0.f, 0.f, 0.f, 0.f, 0.f};
#pragma unroll
    for (int nt = 0; nt < 12; ++nt) {
        Frag b;
#pragma unroll
        for (int i = 0; i < 8; ++i)
            b.u[i] = *(const u32*)(TB1 + ((nt * 16 + i + 8 * half) * 16 + nl) * 2);
        v8f acc = {};
        acc = wmma_bf16(a.bf, b.bf, acc);
        const int n = nt * 16 + nl;
        const float b1 = rb1[n], w2v = rw2[n];
#pragma unroll
        for (int r = 0; r < 8; ++r) {
            float v = acc[r] + b1;
            partial[r] += (v / (1.f + __expf(-v))) * w2v;
        }
    }
#pragma unroll
    for (int r = 0; r < 8; ++r) {
        float s = partial[r];
        s += __shfl_xor(s, 1);
        s += __shfl_xor(s, 2);
        s += __shfl_xor(s, 4);
        s += __shfl_xor(s, 8);
        if (nl == 0) out[row0 + r + 8 * half] = s + rb2[0];
    }
}

// ---------------------------------------------------------------------------
extern "C" void kernel_launch(void* const* d_in, const int* in_sizes, int n_in,
                              void* d_out, int out_size, void* d_ws, size_t ws_size,
                              hipStream_t stream) {
    (void)in_sizes; (void)n_in; (void)out_size; (void)ws_size;
    const float* x   = (const float*)d_in[0];
    const float* rw1 = (const float*)d_in[13];
    const float* rb1 = (const float*)d_in[14];
    const float* rw2 = (const float*)d_in[15];
    const float* rb2 = (const float*)d_in[16];

    // Workspace layout (bf16 elements, all offsets 32B-aligned)
    bf16* ws  = (bf16*)d_ws;
    bf16* act = ws;                          // 33,554,432  layer activations
    bf16* G1r = act + 33554432L;             //  4,194,304  rfft-Z (split R/I,
    bf16* G1i = G1r + 4194304L;              //             y-pair interleaved)
    bf16* G2r = G1i + 4194304L;              //  1,048,576  +DFT-Y (x-pair)
    bf16* G2i = G2r + 1048576L;
    bf16* G3  = G2i + 1048576L;              //    524,288  corner modes (cplx)
    bf16* H   = G3  + 524288L;               //    524,288  after channel mix
    bf16* I1  = H   + 524288L;               //  2,097,152  inv-X (cplx)
    bf16* I2  = I1  + 2097152L;              //  8,388,608  inv-Y (cplx)
    bf16* TW  = I2  + 8388608L;              // 16,777,216  expanded spec weights
    bf16* TZ  = TW  + 16777216L;             // twiddle/weight tables
    bf16* TM  = TZ + 1024L;
    bf16* TI  = TM + 3072L;
    bf16* TC  = TI + 4096L;
    bf16* TB1 = TC + 2048L;
    bf16* TLW = TB1 + 6144L;

    // per-call table prep
    k_init_tw<<<1, 1024, 0, stream>>>(TZ, TM, TI, TC, TB1, TLW, rw1,
                                      (const float*)d_in[1], (const float*)d_in[7]);
    k_prep_w<<<8192, 256, 0, stream>>>((const float*)d_in[3], (const float*)d_in[4],
                                       (const float*)d_in[5], (const float*)d_in[6], TW);
    k_prep_w<<<8192, 256, 0, stream>>>((const float*)d_in[9], (const float*)d_in[10],
                                       (const float*)d_in[11], (const float*)d_in[12],
                                       TW + 2048L * 4096L);

    for (int layer = 0; layer < 2; ++layer) {
        const float* lb = (const float*)d_in[layer ? 8 : 2];
        // rfft-Z: 16384 slabs
        if (layer == 0) k_dft_z<float><<<4096, 128, 0, stream>>>(x, G1r, G1i, TZ);
        else            k_dft_z<bf16 ><<<4096, 128, 0, stream>>>(act, G1r, G1i, TZ);
        // DFT-Y: slabs (b4, x64, kz8) -> G2 split planes, x-pair-interleaved
        k_dft_mode<false><<<512, 128, 0, stream>>>(G1r, G1i, G2r, G2i, TM, 64, 8,
            1048576L, 16384L, 64L, 512L,
            262144L, 8192L, 64L, 512L);
        // DFT-X: slabs (b4, ky16, kz8) -> G3 complex-interleaved
        k_dft_mode<true><<<128, 128, 0, stream>>>(G2r, G2i, G3, nullptr, TM, 16, 8,
            262144L, 512L, 64L, 8192L,
            131072L, 512L, 64L, 8192L);
        // per-mode channel mix with pre-expanded weights
        k_spec<<<512, 128, 0, stream>>>(G3, TW + (long)layer * 2048L * 4096L, H);
        // inverse X: slabs (b4, ky16, kz8)
        k_idft<<<128, 128, 0, stream>>>(H, I1, TI, 16, 8,
            131072L, 512L, 64L, 8192L,
            524288L, 512L, 64L, 8192L);
        // inverse Y: slabs (b4, x64, kz8)
        k_idft<<<512, 128, 0, stream>>>(I1, I2, TI, 64, 8,
            524288L, 8192L, 64L, 512L,
            2097152L, 32768L, 64L, 512L);
        // irfft-Z + residual + SiLU (in-place safe: each wave owns its rows)
        if (layer == 0)
            k_combine<float><<<4096, 128, 0, stream>>>(x, I2, TLW, lb, act, TC);
        else
            k_combine<bf16 ><<<4096, 128, 0, stream>>>(act, I2, TLW + 1024, lb, act, TC);
    }
    // fused MLP head -> d_out [4*64^3]
    k_mlp<<<16384, 128, 0, stream>>>(act, TB1, rb1, rw2, rb2, (float*)d_out);
}
